// GPT_74053826117762
// MI455X (gfx1250) — compile-verified, matrix operand order checked by reading, not run on hardware
//
#include <hip/hip_runtime.h>
#include <hip/hip_bf16.h>
#include <cstdint>

// ---------------------------------------------------------------------------
// GPT forward pass for gfx1250 (MI455X): f32 residual stream, f16 WMMA GEMMs
// (v_wmma_f32_16x16x32_f16), double-buffered LDS with CDNA5 async global->LDS
// copies (global_load_async_to_lds_b128 / s_wait_asynccnt), flash-attention
// with online softmax.
// ---------------------------------------------------------------------------

typedef __attribute__((ext_vector_type(16))) _Float16 v16h;
typedef __attribute__((ext_vector_type(8)))  _Float16 v8h;
typedef __attribute__((ext_vector_type(8)))  float    v8f;
typedef __attribute__((ext_vector_type(4)))  float    v4f;

#define GPT_L  8
#define GPT_E  1024
#define GPT_NH 16
#define GPT_HD 64
#define GPT_V  32000
#define GPT_T  1024
#define GPT_B  2
#define GPT_FF 4096
#define GPT_M  (GPT_B * GPT_T)   // 2048 rows of the activation matrix

__device__ __forceinline__ v8f wmma16(v16h a, v16h b, v8f c) {
  // D = A(16x32 f16) * B(32x16 f16) + C(16x16 f32)
  return __builtin_amdgcn_wmma_f32_16x16x32_f16(
      /*neg_a=*/false, a, /*neg_b=*/false, b,
      /*c_mod=*/(short)0, c, /*reuse_a=*/false, /*reuse_b=*/false);
}

// CDNA5 async global->LDS copy, 16 bytes per lane.  Tracked by ASYNCcnt.
// Generic-pointer low 32 bits of a __shared__ address are the wave-relative
// LDS byte offset (flat aperture rule).
__device__ __forceinline__ void async_copy_b128(const _Float16* gsrc,
                                                _Float16* lds_dst) {
  const unsigned lds_off = (unsigned)(uintptr_t)lds_dst;
  const unsigned long long gaddr = (unsigned long long)(uintptr_t)gsrc;
  asm volatile("global_load_async_to_lds_b128 %0, %1, off"
               :: "v"(lds_off), "v"(gaddr)
               : "memory");
}
__device__ __forceinline__ void wait_async() {
  asm volatile("s_wait_asynccnt 0x0" ::: "memory");
}

// Load a 16x32 WMMA operand fragment from an LDS row (row-major layout).
// ISA layout (16-bit A/B): lane L -> row (L&15); kb = (L>>4)*8;
// elements 0..7  = K[koff+kb   .. koff+kb+7]
// elements 8..15 = K[koff+16+kb.. koff+16+kb+7]
__device__ __forceinline__ v16h ld_frag(const _Float16* row, int koff, int kb) {
  const v8h lo = *(const v8h*)(row + koff + kb);
  const v8h hi = *(const v8h*)(row + koff + kb + 16);
  v16h r;
#pragma unroll
  for (int i = 0; i < 8; ++i) { r[i] = lo[i]; r[i + 8] = hi[i]; }
  return r;
}

__device__ __forceinline__ float gelu_exact(float v) {
  return 0.5f * v * (1.0f + erff(v * 0.70710678118654752f));
}

// ---------------------------------------------------------------------------
// Embedding: x[row, :] = tok_emb[idx[row], :] + pos_emb[row % T, :]
// ---------------------------------------------------------------------------
__global__ __launch_bounds__(256) void embed_kernel(
    const int* __restrict__ idx, const float* __restrict__ tok,
    const float* __restrict__ pos, float* __restrict__ x) {
  const int row = blockIdx.x;
  const int t = row % GPT_T;
  const int token = idx[row];
  const int c = threadIdx.x * 4;
  const float* tp = tok + (long long)token * GPT_E + c;
  const float* pp = pos + (long long)t * GPT_E + c;
  float* xp = x + (long long)row * GPT_E + c;
#pragma unroll
  for (int i = 0; i < 4; ++i) xp[i] = tp[i] + pp[i];
}

// ---------------------------------------------------------------------------
// LayerNorm (biased variance, eps 1e-5), fp32 in -> f16 out
// ---------------------------------------------------------------------------
__global__ __launch_bounds__(256) void ln_kernel(
    const float* __restrict__ x, const float* __restrict__ gamma,
    const float* __restrict__ beta, _Float16* __restrict__ out) {
  const int row = blockIdx.x;
  const int tid = threadIdx.x;
  const int c0 = tid * 4;
  const float* xr = x + (long long)row * GPT_E;
  float v[4], s = 0.f, sq = 0.f;
#pragma unroll
  for (int i = 0; i < 4; ++i) { v[i] = xr[c0 + i]; s += v[i]; sq += v[i] * v[i]; }
  __shared__ float rs[256], rq[256];
  rs[tid] = s; rq[tid] = sq;
  __syncthreads();
  for (int off = 128; off > 0; off >>= 1) {
    if (tid < off) { rs[tid] += rs[tid + off]; rq[tid] += rq[tid + off]; }
    __syncthreads();
  }
  const float mean = rs[0] * (1.0f / GPT_E);
  const float var = rq[0] * (1.0f / GPT_E) - mean * mean;
  const float rstd = rsqrtf(var + 1e-5f);
  _Float16* op = out + (long long)row * GPT_E;
#pragma unroll
  for (int i = 0; i < 4; ++i)
    op[c0 + i] = (_Float16)((v[i] - mean) * rstd * gamma[c0 + i] + beta[c0 + i]);
}

// ---------------------------------------------------------------------------
// Tiled WMMA GEMM:  C[M,N] = A[M,K](f16) * W[K,N](f32->f16) (+bias)(+resid)...
// Block tile 128x128, 8 waves of 32x64 each (2x4 WMMA accumulators), K-step 32.
// Double-buffered LDS: A tile via async global->LDS (no VGPR staging),
// W tile via fp32 float4 loads converted/transposed through registers.
// One barrier per K-step.
// ---------------------------------------------------------------------------
template <bool OUTH, bool BIAS, bool RESID, bool GELU>
__global__ __launch_bounds__(256) void gemm_kernel(
    const _Float16* __restrict__ A, const float* __restrict__ W,
    const float* __restrict__ bias, const float* __restrict__ resid,
    float* __restrict__ outf, _Float16* __restrict__ outh,
    int M, int N, int K, float scale) {
  __shared__ _Float16 As[2][128][40];   // [buf][row][k]   rows 16B-aligned
  __shared__ _Float16 Bs[2][128][40];   // [buf][n][k]     N-major weight tile

  const int tid  = threadIdx.x;
  const int lane = tid & 31;
  const int wave = tid >> 5;
  const int wm = wave >> 1;            // 0..3  (wave's 32-row group)
  const int wn = wave & 1;             // 0..1  (wave's 64-col group)
  const int hi = (lane >> 4) & 1;
  const int ln = lane & 15;
  const int kb = hi * 8;
  const int gm0 = blockIdx.y * 128;
  const int gn0 = blockIdx.x * 128;

  const v8f zero8 = {0.f, 0.f, 0.f, 0.f, 0.f, 0.f, 0.f, 0.f};
  v8f acc[2][4];
#pragma unroll
  for (int mt = 0; mt < 2; ++mt)
#pragma unroll
    for (int nt = 0; nt < 4; ++nt) acc[mt][nt] = zero8;

  const int bk = tid >> 5;             // 0..7
  const int bn = (tid & 31) * 4;       // 0..124

  // A tile: 128 rows x 32 halfs = 512 x b128 chunks; 2 per thread, async.
  auto issue_A = [&](int buf, int k0) {
#pragma unroll
    for (int c = 0; c < 2; ++c) {
      const int chunk = tid * 2 + c;
      const int row = chunk >> 2;
      const int col = (chunk & 3) * 8;
      async_copy_b128(A + (long long)(gm0 + row) * K + k0 + col,
                      &As[buf][row][col]);
    }
  };
  // W tile: 32(K) x 128(N) fp32, coalesced float4 loads into registers.
  auto load_B = [&](v4f* breg, int k0) {
#pragma unroll
    for (int it = 0; it < 4; ++it)
      breg[it] = *(const v4f*)(W + (long long)(k0 + bk + it * 8) * N + gn0 + bn);
  };
  auto store_B = [&](int buf, const v4f* breg) {
#pragma unroll
    for (int it = 0; it < 4; ++it) {
      const int kk = bk + it * 8;
      Bs[buf][bn + 0][kk] = (_Float16)breg[it][0];
      Bs[buf][bn + 1][kk] = (_Float16)breg[it][1];
      Bs[buf][bn + 2][kk] = (_Float16)breg[it][2];
      Bs[buf][bn + 3][kk] = (_Float16)breg[it][3];
    }
  };

  // Prologue: stage tile 0.
  {
    issue_A(0, 0);
    v4f breg[4];
    load_B(breg, 0);
    store_B(0, breg);
    wait_async();
    __syncthreads();
  }

  int cur = 0;
  for (int k0 = 0; k0 < K; k0 += 32) {
    const int nxt = cur ^ 1;
    const bool more = (k0 + 32) < K;
    v4f breg[4];
    if (more) {
      issue_A(nxt, k0 + 32);     // async -> hidden under the WMMAs below
      load_B(breg, k0 + 32);     // global fp32 loads in flight during compute
    }
    // Compute from the current buffers: 8 WMMAs per wave per K-step.
    const v16h a0 = ld_frag(&As[cur][wm * 32 + ln][0], 0, kb);
    const v16h a1 = ld_frag(&As[cur][wm * 32 + 16 + ln][0], 0, kb);
#pragma unroll
    for (int nt = 0; nt < 4; ++nt) {
      const v16h b = ld_frag(&Bs[cur][wn * 64 + nt * 16 + ln][0], 0, kb);
      acc[0][nt] = wmma16(a0, b, acc[0][nt]);
      acc[1][nt] = wmma16(a1, b, acc[1][nt]);
    }
    if (more) {
      store_B(nxt, breg);
      wait_async();              // As[nxt] fully landed in LDS
    }
    __syncthreads();             // publish nxt buffers / retire cur readers
    cur = nxt;
  }

  // Epilogue: C/D layout -> element (VGPR r, lane) = row r+8*(lane>>4), col lane&15
#pragma unroll
  for (int mt = 0; mt < 2; ++mt) {
#pragma unroll
    for (int nt = 0; nt < 4; ++nt) {
#pragma unroll
      for (int r = 0; r < 8; ++r) {
        const int row_g = gm0 + wm * 32 + mt * 16 + r + 8 * hi;
        const int col_g = gn0 + wn * 64 + nt * 16 + ln;
        float v = acc[mt][nt][r] * scale;
        if (BIAS) v += bias[col_g];
        if (GELU) v = gelu_exact(v);
        const long long oi = (long long)row_g * N + col_g;
        if (RESID)      outf[oi] = resid[oi] + v;
        else if (OUTH)  outh[oi] = (_Float16)v;
        else            outf[oi] = v;
      }
    }
  }
}

// ---------------------------------------------------------------------------
// Flash attention: one block = 64 queries of one (b, h); 4 waves x 16 queries.
// Online softmax; S = Q*K^T and O += P*V via WMMA. Scale folded into q.
// Q/K tiles arrive via async global->LDS; V is transposed through VGPRs.
// ---------------------------------------------------------------------------
__global__ __launch_bounds__(128) void attn_kernel(
    const _Float16* __restrict__ q, const _Float16* __restrict__ k,
    const _Float16* __restrict__ v, _Float16* __restrict__ o) {
  const int qb = blockIdx.x;   // query block (64 rows)
  const int h  = blockIdx.y;
  const int b  = blockIdx.z;
  const int tid  = threadIdx.x;
  const int wave = tid >> 5;
  const int lane = tid & 31;
  const int hi = (lane >> 4) & 1;
  const int ln = lane & 15;
  const int kb = hi * 8;

  __shared__ _Float16 Qs[64][72];       // [query][d]
  __shared__ _Float16 Ks[64][72];       // [key][d]   (== B^T layout for Q*K^T)
  __shared__ _Float16 VTs[64][72];      // [d][key]   (== B^T layout for P*V)
  __shared__ _Float16 Ps[4][16][72];    // per-wave P tile [qrow][key]

  // Q tile: 64x64 halfs = 512 b128 chunks; 4 per thread, async.
  const long long base_q = ((long long)b * GPT_T + qb * 64) * GPT_E + h * GPT_HD;
#pragma unroll
  for (int c = 0; c < 4; ++c) {
    const int chunk = tid * 4 + c;
    const int row = chunk >> 3;
    const int col = (chunk & 7) * 8;
    async_copy_b128(q + base_q + (long long)row * GPT_E + col, &Qs[row][col]);
  }

  const v8f zero8 = {0.f, 0.f, 0.f, 0.f, 0.f, 0.f, 0.f, 0.f};
  v8f O[4]; O[0] = zero8; O[1] = zero8; O[2] = zero8; O[3] = zero8;
  float mrow[8], lrow[8];
#pragma unroll
  for (int r = 0; r < 8; ++r) { mrow[r] = -3.0e38f; lrow[r] = 0.f; }

  for (int j = 0; j <= qb; ++j) {
    __syncthreads();  // protect Ks/VTs from previous iteration's readers
    const long long base_k = ((long long)b * GPT_T + j * 64) * GPT_E + h * GPT_HD;
    // K tile: async (natural layout is already B^T for Q*K^T).
#pragma unroll
    for (int c = 0; c < 4; ++c) {
      const int chunk = tid * 4 + c;
      const int row = chunk >> 3;
      const int col = (chunk & 7) * 8;
      async_copy_b128(k + base_k + (long long)row * GPT_E + col, &Ks[row][col]);
    }
    // V tile: VGPR path with transpose into VTs[d][key].
#pragma unroll
    for (int i = 0; i < 4; ++i) {
      const int li = tid * 8 + i * 1024;
      const int row = li >> 6, col = li & 63;
      const v8h vv = *(const v8h*)(v + base_k + (long long)row * GPT_E + col);
#pragma unroll
      for (int e = 0; e < 8; ++e) VTs[col + e][row] = vv[e];
    }
    wait_async();    // Q (first pass) + K landed in LDS
    __syncthreads();

    // S = Q * K^T  (16 queries x 64 keys per wave)
    v8f S[4]; S[0] = zero8; S[1] = zero8; S[2] = zero8; S[3] = zero8;
#pragma unroll
    for (int kk = 0; kk < 64; kk += 32) {
      const v16h a = ld_frag(&Qs[wave * 16 + ln][0], kk, kb);
#pragma unroll
      for (int nt = 0; nt < 4; ++nt) {
        const v16h bf = ld_frag(&Ks[nt * 16 + ln][0], kk, kb);
        S[nt] = wmma16(a, bf, S[nt]);
      }
    }

    // causal mask + online softmax update
#pragma unroll
    for (int r = 0; r < 8; ++r) {
      const int qg = qb * 64 + wave * 16 + r + 8 * hi;
#pragma unroll
      for (int nt = 0; nt < 4; ++nt) {
        const int kg = j * 64 + nt * 16 + ln;
        if (kg > qg) S[nt][r] = -3.0e38f;
      }
      float mx = fmaxf(fmaxf(S[0][r], S[1][r]), fmaxf(S[2][r], S[3][r]));
#pragma unroll
      for (int off = 1; off < 16; off <<= 1)
        mx = fmaxf(mx, __shfl_xor(mx, off, 32));
      const float mnew = fmaxf(mrow[r], mx);
      const float alpha = __expf(mrow[r] - mnew);
      mrow[r] = mnew;
      float psum = 0.f;
#pragma unroll
      for (int nt = 0; nt < 4; ++nt) {
        const float p = __expf(S[nt][r] - mnew);
        Ps[wave][r + 8 * hi][nt * 16 + ln] = (_Float16)p;
        psum += p;
      }
#pragma unroll
      for (int off = 1; off < 16; off <<= 1)
        psum += __shfl_xor(psum, off, 32);
      lrow[r] = lrow[r] * alpha + psum;
#pragma unroll
      for (int a = 0; a < 4; ++a) O[a][r] *= alpha;
    }

    // O += P * V   (P read back through LDS in A-fragment layout; same-wave
    // LDS RAW is safe: DS ops from one wave are processed in order)
#pragma unroll
    for (int kk = 0; kk < 64; kk += 32) {
      const v16h a = ld_frag(&Ps[wave][ln][0], kk, kb);
#pragma unroll
      for (int dt = 0; dt < 4; ++dt) {
        const v16h bf = ld_frag(&VTs[dt * 16 + ln][0], kk, kb);
        O[dt] = wmma16(a, bf, O[dt]);
      }
    }
  }

  // normalize + store
#pragma unroll
  for (int dt = 0; dt < 4; ++dt) {
#pragma unroll
    for (int r = 0; r < 8; ++r) {
      const int qg = qb * 64 + wave * 16 + r + 8 * hi;
      const float val = O[dt][r] / lrow[r];
      o[((long long)b * GPT_T + qg) * GPT_E + h * GPT_HD + dt * 16 + ln] =
          (_Float16)val;
    }
  }
}

// ---------------------------------------------------------------------------
// Host orchestration
// ---------------------------------------------------------------------------
extern "C" void kernel_launch(void* const* d_in, const int* in_sizes, int n_in,
                              void* d_out, int out_size, void* d_ws,
                              size_t ws_size, hipStream_t stream) {
  (void)in_sizes; (void)n_in; (void)out_size; (void)ws_size;

  const int*   idx    = (const int*)d_in[0];
  const float* tok    = (const float*)d_in[1];
  const float* pos    = (const float*)d_in[2];
  const float* Wq     = (const float*)d_in[3];
  const float* Wk     = (const float*)d_in[4];
  const float* Wv     = (const float*)d_in[5];
  const float* Wo     = (const float*)d_in[6];
  const float* bo     = (const float*)d_in[7];
  const float* ln1s   = (const float*)d_in[8];
  const float* ln1b   = (const float*)d_in[9];
  const float* W1     = (const float*)d_in[10];
  const float* b1     = (const float*)d_in[11];
  const float* W2     = (const float*)d_in[12];
  const float* b2     = (const float*)d_in[13];
  const float* ln2s   = (const float*)d_in[14];
  const float* ln2b   = (const float*)d_in[15];
  const float* lnfs   = (const float*)d_in[16];
  const float* lnfb   = (const float*)d_in[17];
  const float* lm_w   = (const float*)d_in[18];
  const float* lm_b   = (const float*)d_in[19];
  float* out = (float*)d_out;

  // Workspace layout (all 256B aligned)
  char* ws = (char*)d_ws;
  float*    x    = (float*)   (ws + 0);                        // 8 MiB
  _Float16* h16  = (_Float16*)(ws + (size_t)8  * 1024 * 1024); // 4 MiB
  _Float16* q16  = (_Float16*)(ws + (size_t)12 * 1024 * 1024); // 4 MiB
  _Float16* k16  = (_Float16*)(ws + (size_t)16 * 1024 * 1024); // 4 MiB
  _Float16* v16  = (_Float16*)(ws + (size_t)20 * 1024 * 1024); // 4 MiB
  _Float16* o16  = (_Float16*)(ws + (size_t)24 * 1024 * 1024); // 4 MiB
  _Float16* ff16 = (_Float16*)(ws + (size_t)28 * 1024 * 1024); // 16 MiB

  const dim3 blk256(256), blk128(128);
  const dim3 gE(GPT_E / 128, GPT_M / 128);    // N=1024 GEMMs
  const dim3 gF(GPT_FF / 128, GPT_M / 128);   // N=4096 GEMM
  const dim3 gV(GPT_V / 128, GPT_M / 128);    // N=32000 GEMM
  const dim3 gA(GPT_T / 64, GPT_NH, GPT_B);   // attention

  embed_kernel<<<GPT_M, blk256, 0, stream>>>(idx, tok, pos, x);

  for (int l = 0; l < GPT_L; ++l) {
    const float* Wq_l = Wq + (size_t)l * GPT_E * GPT_E;
    const float* Wk_l = Wk + (size_t)l * GPT_E * GPT_E;
    const float* Wv_l = Wv + (size_t)l * GPT_E * GPT_E;
    const float* Wo_l = Wo + (size_t)l * GPT_E * GPT_E;
    const float* W1_l = W1 + (size_t)l * GPT_E * GPT_FF;
    const float* W2_l = W2 + (size_t)l * GPT_FF * GPT_E;

    ln_kernel<<<GPT_M, blk256, 0, stream>>>(x, ln1s + l * GPT_E, ln1b + l * GPT_E, h16);

    // q gets the 1/sqrt(HD) scale folded in
    gemm_kernel<true, false, false, false><<<gE, blk256, 0, stream>>>(
        h16, Wq_l, nullptr, nullptr, nullptr, q16, GPT_M, GPT_E, GPT_E, 0.125f);
    gemm_kernel<true, false, false, false><<<gE, blk256, 0, stream>>>(
        h16, Wk_l, nullptr, nullptr, nullptr, k16, GPT_M, GPT_E, GPT_E, 1.0f);
    gemm_kernel<true, false, false, false><<<gE, blk256, 0, stream>>>(
        h16, Wv_l, nullptr, nullptr, nullptr, v16, GPT_M, GPT_E, GPT_E, 1.0f);

    attn_kernel<<<gA, blk128, 0, stream>>>(q16, k16, v16, o16);

    // x = x + o @ Wo + bo
    gemm_kernel<false, true, true, false><<<gE, blk256, 0, stream>>>(
        o16, Wo_l, bo + l * GPT_E, x, x, nullptr, GPT_M, GPT_E, GPT_E, 1.0f);

    ln_kernel<<<GPT_M, blk256, 0, stream>>>(x, ln2s + l * GPT_E, ln2b + l * GPT_E, h16);

    // ff = gelu(h @ W1 + b1)  (f16)
    gemm_kernel<true, true, false, true><<<gF, blk256, 0, stream>>>(
        h16, W1_l, b1 + l * GPT_FF, nullptr, nullptr, ff16, GPT_M, GPT_FF, GPT_E, 1.0f);
    // x = x + ff @ W2 + b2
    gemm_kernel<false, true, true, false><<<gE, blk256, 0, stream>>>(
        ff16, W2_l, b2 + l * GPT_E, x, x, nullptr, GPT_M, GPT_E, GPT_FF, 1.0f);
  }

  ln_kernel<<<GPT_M, blk256, 0, stream>>>(x, lnfs, lnfb, h16);
  // logits = h @ lm_w + lm_b  (fp32 out)
  gemm_kernel<false, true, false, false><<<gV, blk256, 0, stream>>>(
      h16, lm_w, lm_b, nullptr, out, nullptr, GPT_M, GPT_V, GPT_E, 1.0f);
}
